// BuildingVecMoveModel_7533372637728
// MI455X (gfx1250) — matrix operand order repaired and checked by reading, not compiled, not run on hardware
//
#include <hip/hip_runtime.h>
#include <hip/hip_bf16.h>

typedef __attribute__((ext_vector_type(8)))  float  v8f;
typedef __attribute__((ext_vector_type(16))) __bf16 v16bf;

#define NEG_SENTINEL (-3.402823466e38f)

union FragBF {
    v16bf v;
    uint4 u[2];
};

// ---------------------------------------------------------------------------
// Native CDNA5 f32 atomic max (non-returning -> STOREcnt, no VGPR writeback).
// ISA §15.18.3 opcode 82: GLOBAL_ATOMIC_MAX_NUM_F32 (IEEE maximumNumber).
// max() is associative/commutative -> scatter order does not affect result.
// ---------------------------------------------------------------------------
__device__ __forceinline__ void atom_max_f32(float* p, float v) {
    asm volatile("global_atomic_max_num_f32 %0, %1, off scope:SCOPE_DEV"
                 :: "v"(p), "v"(v)
                 : "memory");
}

// ---------------------------------------------------------------------------
// Fill a float buffer with a constant (segment-max accumulator init).
// ---------------------------------------------------------------------------
__global__ void fill_f32(float* __restrict__ p, float v, int n) {
    int i = blockIdx.x * blockDim.x + threadIdx.x;
    if (i < n) p[i] = v;
}

// ---------------------------------------------------------------------------
// Convert f32 -> bf16, optionally mapping "never touched" sentinel to 0
// (reference: where(agg <= NEG, 0, agg)).
// ---------------------------------------------------------------------------
__global__ void cvt_bf16(const float* __restrict__ in, __bf16* __restrict__ out,
                         int n, int clean) {
    int i = blockIdx.x * blockDim.x + threadIdx.x;
    if (i >= n) return;
    float v = in[i];
    if (clean && v <= -3.0e38f) v = 0.0f;
    out[i] = (__bf16)v;
}

// ---------------------------------------------------------------------------
// Pack [Wl ; Wr] (each F x H, row-major) into transposed bf16 [H][2F] so the
// WMMA B-fragment per lane is two K-contiguous 16-byte reads.
// ---------------------------------------------------------------------------
__global__ void pack_w(const float* __restrict__ Wl, const float* __restrict__ Wr,
                       int F, int H, __bf16* __restrict__ out) {
    int idx = blockIdx.x * blockDim.x + threadIdx.x;
    int K2 = 2 * F;
    if (idx >= H * K2) return;
    int n = idx / K2;
    int k = idx - n * K2;
    float v = (k < F) ? Wl[(size_t)k * H + n] : Wr[(size_t)(k - F) * H + n];
    out[idx] = (__bf16)v;
}

// ---------------------------------------------------------------------------
// Edge scatter-max: thread handles 4 channels of one edge. float4 gather from
// x[src] (L2-resident), 4x native f32 atomic-max into agg[dst].
// ---------------------------------------------------------------------------
__global__ void scatter_max(const float* __restrict__ x,
                            const int* __restrict__ src,
                            const int* __restrict__ dst,
                            float* __restrict__ agg,
                            int E, int F, int shift) {
    int t = blockIdx.x * blockDim.x + threadIdx.x;
    int e = t >> shift;
    if (e >= E) return;
    int c = (t & ((1 << shift) - 1)) << 2;
    int s = src[e];
    int d = dst[e];
    float4 v = *(const float4*)(x + (size_t)s * F + c);
    float* p = agg + (size_t)d * F + c;
    atom_max_f32(p + 0, v.x);
    atom_max_f32(p + 1, v.y);
    atom_max_f32(p + 2, v.z);
    atom_max_f32(p + 3, v.w);
}

// ---------------------------------------------------------------------------
// One K-panel of the fused SAGE GEMM: acc += A[N,F] tile  x  Wt cols.
// A frag (ISA §7.12.2, 16-bit A 16x32): lane holds row lane%16,
//   K = k + 8*(lane>=16) + {0..7, 16..23}  -> two 16B loads.
// B frag (32x16): lane holds col lane%16, K = k + 8*(lane>=16) + {0..7, 16..23}
//   -> two 16B loads from the transposed weight pack.
// ---------------------------------------------------------------------------
__device__ __forceinline__ void mma_panel(const __bf16* __restrict__ A, int F,
                                          const __bf16* __restrict__ Wt, int Ktot,
                                          int kbase, int mrow, int col_base,
                                          int l16, int lhalf, v8f acc[4]) {
    const __bf16* ap = A + (size_t)mrow * F + 8 * lhalf;
    const __bf16* bp0 = Wt + (size_t)(col_base + l16) * Ktot + kbase + 8 * lhalf;
    for (int k = 0; k < F; k += 32) {
        FragBF a;
        a.u[0] = *(const uint4*)(ap);
        a.u[1] = *(const uint4*)(ap + 16);
        ap += 32;
#pragma unroll
        for (int t = 0; t < 4; ++t) {
            FragBF b;
            const __bf16* bp = bp0 + (size_t)(t * 16) * Ktot + k;
            b.u[0] = *(const uint4*)(bp);
            b.u[1] = *(const uint4*)(bp + 16);
            acc[t] = __builtin_amdgcn_wmma_f32_16x16x32_bf16(
                false, a.v, false, b.v, (short)0, acc[t], false, false);
        }
    }
}

// ---------------------------------------------------------------------------
// Fused SAGE GEMM: out = relu([A0 | A1] @ Wt^T + bias). Block = 8 waves; each
// wave computes a 16x64 strip (1 A-frag amortized over 4 WMMAs per K-step).
// Optionally mirrors the f32 result to bf16 (next layer's WMMA A operand).
// ---------------------------------------------------------------------------
__global__ __launch_bounds__(256)
void gemm_sage(const __bf16* __restrict__ A0, const __bf16* __restrict__ A1, int F,
               const __bf16* __restrict__ Wt,
               const float* __restrict__ bias0, const float* __restrict__ bias1,
               int Hsplit, float* __restrict__ out, __bf16* __restrict__ outbf,
               int N, int H) {
    const int lane  = threadIdx.x & 31;
    const int wave  = threadIdx.x >> 5;
    const int l16   = lane & 15;
    const int lhalf = lane >> 4;
    const int row_base = blockIdx.x * 128 + wave * 16;
    const int col_base = blockIdx.y * 64;
    const int Ktot = 2 * F;

    // Branchless clamp so OOB-row waves still feed WMMA with valid memory
    // (EXEC stays all-ones through the whole K loop).
    int mrow = row_base + l16;
    if (mrow > N - 1) mrow = N - 1;

    v8f acc[4];
#pragma unroll
    for (int t = 0; t < 4; ++t) acc[t] = 0.0f;

    mma_panel(A0, F, Wt, Ktot, 0, mrow, col_base, l16, lhalf, acc);  // agg panel
    mma_panel(A1, F, Wt, Ktot, F, mrow, col_base, l16, lhalf, acc);  // self panel

#pragma unroll
    for (int t = 0; t < 4; ++t) {
        int n = col_base + t * 16 + l16;
        float b = (n < Hsplit) ? bias0[n] : bias1[n - Hsplit];
#pragma unroll
        for (int j = 0; j < 8; ++j) {
            int r = row_base + j + 8 * lhalf;
            if (r < N) {
                float v = acc[t][j] + b;
                v = v > 0.0f ? v : 0.0f;
                out[(size_t)r * H + n] = v;
                if (outbf) outbf[(size_t)r * H + n] = (__bf16)v;
            }
        }
    }
}

// ---------------------------------------------------------------------------
// Heads: one wave per node; rtmv is [N,512] = [rt_hidden | mv_hidden].
// rt = row[0:256].Wa + ba ; mv = row[256:512].Wm + bm
// ---------------------------------------------------------------------------
__global__ void head_dot(const float* __restrict__ rtmv,
                         const float* __restrict__ Wa, const float* __restrict__ ba,
                         const float* __restrict__ Wm, const float* __restrict__ bm,
                         float* __restrict__ out, int N) {
    int gid  = blockIdx.x * blockDim.x + threadIdx.x;
    int node = gid >> 5;
    int lane = threadIdx.x & 31;
    if (node >= N) return;
    const float* row = rtmv + (size_t)node * 512;
    float srt = 0.0f, smv = 0.0f;
#pragma unroll
    for (int j = 0; j < 8; ++j) {
        int c = lane + 32 * j;
        srt += row[c] * Wa[c];
        smv += row[256 + c] * Wm[c];
    }
#pragma unroll
    for (int o = 16; o > 0; o >>= 1) {
        srt += __shfl_xor(srt, o, 32);
        smv += __shfl_xor(smv, o, 32);
    }
    if (lane == 0) {
        out[node]     = srt + ba[0];
        out[N + node] = smv + bm[0];
    }
}

// ---------------------------------------------------------------------------
extern "C" void kernel_launch(void* const* d_in, const int* in_sizes, int n_in,
                              void* d_out, int out_size, void* d_ws, size_t ws_size,
                              hipStream_t stream) {
    (void)in_sizes; (void)n_in; (void)out_size; (void)ws_size;
    const int N = 50000, E = 800000;

    const float* x    = (const float*)d_in[0];
    const int*   ei   = (const int*)d_in[1];   // [2,E] int32 (JAX x64 disabled)
    const int*   src  = ei;
    const int*   dst  = ei + E;
    const float* Wl1  = (const float*)d_in[2];
    const float* bl1  = (const float*)d_in[3];
    const float* Wr1  = (const float*)d_in[4];
    const float* Wl2  = (const float*)d_in[5];
    const float* bl2  = (const float*)d_in[6];
    const float* Wr2  = (const float*)d_in[7];
    const float* Wla  = (const float*)d_in[8];
    const float* bla  = (const float*)d_in[9];
    const float* Wra  = (const float*)d_in[10];
    const float* Wa   = (const float*)d_in[11];
    const float* ba   = (const float*)d_in[12];
    const float* Wlm  = (const float*)d_in[13];
    const float* blm  = (const float*)d_in[14];
    const float* Wrm  = (const float*)d_in[15];
    const float* Wm   = (const float*)d_in[16];
    const float* bm   = (const float*)d_in[17];
    float* out = (float*)d_out;

    // Workspace arena. bufA and bufB are contiguous so they double as the
    // [N,512] rt|mv GEMM output once agg/h1 are dead.
    float*  bufA  = (float*)d_ws;                     // agg f32 / rtmv rows 0..N/2-1
    float*  bufB  = bufA + (size_t)N * 256;           // h1 f32  / rtmv rows N/2..N-1
    float*  h2    = bufB + (size_t)N * 256;           // h2 f32
    __bf16* xbf   = (__bf16*)(h2 + (size_t)N * 256);  // x bf16     [N,128]
    __bf16* h1bf  = xbf  + (size_t)N * 128;           // h1 bf16    [N,256]
    __bf16* h2bf  = h1bf + (size_t)N * 256;           // h2 bf16    [N,256]
    __bf16* aggbf = h2bf + (size_t)N * 256;           // agg bf16   [N,256]
    __bf16* Wt1   = aggbf + (size_t)N * 256;          // [256][256]
    __bf16* Wt2   = Wt1 + 256 * 256;                  // [256][512]
    __bf16* Wt3   = Wt2 + 256 * 512;                  // [512][512] (rt 0..255, mv 256..511)

    const int T = 256;
    dim3 gemm_grid((N + 127) / 128, 4);
    dim3 gemm_grid3((N + 127) / 128, 8);

    // --- weight packing (bf16, transposed, [Wl;Wr] fused K) ---
    pack_w<<<(256 * 256 + T - 1) / T, T, 0, stream>>>(Wl1, Wr1, 128, 256, Wt1);
    pack_w<<<(256 * 512 + T - 1) / T, T, 0, stream>>>(Wl2, Wr2, 256, 256, Wt2);
    pack_w<<<(256 * 512 + T - 1) / T, T, 0, stream>>>(Wla, Wra, 256, 256, Wt3);
    pack_w<<<(256 * 512 + T - 1) / T, T, 0, stream>>>(Wlm, Wrm, 256, 256, Wt3 + 256 * 512);
    cvt_bf16<<<(N * 128 + T - 1) / T, T, 0, stream>>>(x, xbf, N * 128, 0);

    // --- layer 1: agg = segmax(x), h1 = relu([agg|x] @ Wt1 + bl1) ---
    fill_f32<<<(N * 128 + T - 1) / T, T, 0, stream>>>(bufA, NEG_SENTINEL, N * 128);
    scatter_max<<<((E << 5) + T - 1) / T, T, 0, stream>>>(x, src, dst, bufA, E, 128, 5);
    cvt_bf16<<<(N * 128 + T - 1) / T, T, 0, stream>>>(bufA, aggbf, N * 128, 1);
    gemm_sage<<<gemm_grid, T, 0, stream>>>(aggbf, xbf, 128, Wt1, bl1, bl1, 256,
                                           bufB, h1bf, N, 256);

    // --- layer 2: agg = segmax(h1), h2 = relu([agg|h1] @ Wt2 + bl2) ---
    fill_f32<<<(N * 256 + T - 1) / T, T, 0, stream>>>(bufA, NEG_SENTINEL, N * 256);
    scatter_max<<<((E << 6) + T - 1) / T, T, 0, stream>>>(bufB, src, dst, bufA, E, 256, 6);
    cvt_bf16<<<(N * 256 + T - 1) / T, T, 0, stream>>>(bufA, aggbf, N * 256, 1);
    gemm_sage<<<gemm_grid, T, 0, stream>>>(aggbf, h1bf, 256, Wt2, bl2, bl2, 256,
                                           h2, h2bf, N, 256);

    // --- layer 3 (rt & mv share segmax(h2); fused as one H=512 GEMM) ---
    fill_f32<<<(N * 256 + T - 1) / T, T, 0, stream>>>(bufA, NEG_SENTINEL, N * 256);
    scatter_max<<<((E << 6) + T - 1) / T, T, 0, stream>>>(h2, src, dst, bufA, E, 256, 6);
    cvt_bf16<<<(N * 256 + T - 1) / T, T, 0, stream>>>(bufA, aggbf, N * 256, 1);
    gemm_sage<<<gemm_grid3, T, 0, stream>>>(aggbf, h2bf, 256, Wt3, bla, blm, 256,
                                            bufA, (__bf16*)nullptr, N, 512);

    // --- heads ---
    head_dot<<<(N * 32 + T - 1) / T, T, 0, stream>>>(bufA, Wa, ba, Wm, bm, out, N);
}
